// SelfAttention_661424964251
// MI455X (gfx1250) — matrix operand order repaired and checked by reading, not compiled, not run on hardware
//
#include <hip/hip_runtime.h>
#include <math.h>

// ---------------- problem constants ----------------
#define DIMC   1024
#define HEADS  16
#define HD     64
#define BATCH  4
#define NTOK   1024
#define LCOND  77
#define LPAD   96                 // cond length padded to multiple of 32
#define KPADT  (NTOK + LPAD)      // 1120 total keys, 35 chunks of 32
#define EPSV   1e-6f
#define MAX_LOG_SCALE 4.6051701859880914f   // log(100)
#define LOG_PI        1.1447298858494002f   // log(pi)
#define FREQ_STEP     0.0089944730261390f   // log(10)/256
#define QLOG_HD       1.0397207708399179f   // 0.25*log(64)

static_assert(KPADT % 32 == 0, "KV length must be chunkable by 32");

typedef _Float16 v16h __attribute__((ext_vector_type(16)));
typedef _Float16 v8h  __attribute__((ext_vector_type(8)));
typedef float    v8f  __attribute__((ext_vector_type(8)));

__device__ __forceinline__ v8f wmma16(v16h a, v16h b, v8f c) {
  // D = A(16x32 f16) * B(32x16 f16) + C(16x16 f32)
  return __builtin_amdgcn_wmma_f32_16x16x32_f16(false, a, false, b, (short)0, c,
                                                false, false);
}

// A-fragment (16x32 f16): lane holds row m = lane&15.
// half e: k = ((lane>>4)<<4) + e  -> 16 contiguous halves.
__device__ __forceinline__ v16h load_a_frag(const _Float16* rowk, int hi) {
  const _Float16* p = rowk + (hi << 4);
  v8h lo = *(const v8h*)(p);
  v8h hh = *(const v8h*)(p + 8);
  v16h r;
#pragma unroll
  for (int i = 0; i < 8; ++i) { r[i] = lo[i]; r[i + 8] = hh[i]; }
  return r;
}
// B-fragment (32x16 f16): lane holds col n = lane&15, halves are 16 consecutive k
// starting at k = 16*(lane>>4)  -> one contiguous 32-byte load.

// ---------------- small prep kernels ----------------
__global__ void cvt_f32_f16_kernel(const float* __restrict__ s,
                                   _Float16* __restrict__ d, long n) {
  long i = (long)blockIdx.x * 256 + threadIdx.x;
  if (i < n) d[i] = (_Float16)s[i];
}

__global__ void build_cond16_kernel(const float* __restrict__ src,
                                    _Float16* __restrict__ dst) {
  long i = (long)blockIdx.x * 256 + threadIdx.x;   // over B*LPAD*DIM
  if (i >= (long)BATCH * LPAD * DIMC) return;
  int c  = (int)(i & (DIMC - 1));
  long rl = i >> 10;
  int lc = (int)(rl % LPAD);
  int b  = (int)(rl / LPAD);
  dst[i] = (lc < LCOND) ? (_Float16)src[((size_t)b * LCOND + lc) * DIMC + c]
                        : (_Float16)0.f;
}

__global__ void rmsnorm_x_kernel(const float* __restrict__ x,
                                 const float* __restrict__ scale,
                                 _Float16* __restrict__ xn) {
  __shared__ float red[8];
  const int row = blockIdx.x;                 // B*N rows
  const int t = threadIdx.x;                  // 256 threads
  const float* xr = x + (size_t)row * DIMC;
  float v[4]; float s = 0.f;
#pragma unroll
  for (int i = 0; i < 4; ++i) { v[i] = xr[t + i * 256]; s += v[i] * v[i]; }
#pragma unroll
  for (int m = 16; m >= 1; m >>= 1) s += __shfl_xor(s, m, 32);
  if ((t & 31) == 0) red[t >> 5] = s;
  __syncthreads();
  float tot = 0.f;
#pragma unroll
  for (int i = 0; i < 8; ++i) tot += red[i];
  const float r = rsqrtf(tot * (1.0f / DIMC) + EPSV);
  _Float16* orow = xn + (size_t)row * DIMC;
#pragma unroll
  for (int i = 0; i < 4; ++i)
    orow[t + i * 256] = (_Float16)(v[i] * scale[t + i * 256] * r);
}

__global__ void fill_bias_kernel(const unsigned char* __restrict__ mask,
                                 float* __restrict__ bias) {
  int i = blockIdx.x * 256 + threadIdx.x;     // B*KPADT
  if (i >= BATCH * KPADT) return;
  int b = i / KPADT, key = i % KPADT;
  float v = -1e9f;
  if (key < NTOK) v = 0.f;
  else { int lc = key - NTOK; if (lc < LCOND && mask[b * LCOND + lc]) v = 0.f; }
  bias[i] = v;
}

// ------- WMMA GEMM: Out(MxN) = A(MxK) @ W(NxK)^T (+bias) -------
// Wave tile 32x64 (2 A-frags x 4 B-frags -> 8 wmma/k-step), software-pipelined
// double buffering so next-k loads overlap the current wmma group.
__global__ void __launch_bounds__(32)
gemm_wmma_kernel(const _Float16* __restrict__ A, const _Float16* __restrict__ W,
                 float* __restrict__ Out, const float* __restrict__ bias,
                 int M, int Nn, int K) {
  const int lane = threadIdx.x & 31;
  const int ln = lane & 15, hi = lane >> 4;
  const int tilesN = Nn >> 6;
  const int m0 = (blockIdx.x / tilesN) << 5;   // 32-row tile
  const int n0 = (blockIdx.x % tilesN) << 6;   // 64-col tile

  const _Float16* ar0 = A + (size_t)(m0 + ln) * K;
  const _Float16* ar1 = A + (size_t)(m0 + 16 + ln) * K;
  const _Float16* wr  = W + (size_t)(n0 + ln) * K + (hi << 4);

  v8f acc0[4] = {}, acc1[4] = {};
  // prologue: fragments for k0 = 0
  v16h a0 = load_a_frag(ar0, hi);
  v16h a1 = load_a_frag(ar1, hi);
  v16h b0 = *(const v16h*)(wr);
  v16h b1 = *(const v16h*)(wr + (size_t)16 * K);
  v16h b2 = *(const v16h*)(wr + (size_t)32 * K);
  v16h b3 = *(const v16h*)(wr + (size_t)48 * K);

  for (int k0 = 0; k0 < K - 32; k0 += 32) {
    const int kn = k0 + 32;
    __builtin_prefetch(ar0 + kn + 128, 0, 1);     // global_prefetch_b8
    __builtin_prefetch(ar1 + kn + 128, 0, 1);
    v16h a0n = load_a_frag(ar0 + kn, hi);
    v16h a1n = load_a_frag(ar1 + kn, hi);
    v16h b0n = *(const v16h*)(wr + kn);
    v16h b1n = *(const v16h*)(wr + (size_t)16 * K + kn);
    v16h b2n = *(const v16h*)(wr + (size_t)32 * K + kn);
    v16h b3n = *(const v16h*)(wr + (size_t)48 * K + kn);
    acc0[0] = wmma16(a0, b0, acc0[0]); acc1[0] = wmma16(a1, b0, acc1[0]);
    acc0[1] = wmma16(a0, b1, acc0[1]); acc1[1] = wmma16(a1, b1, acc1[1]);
    acc0[2] = wmma16(a0, b2, acc0[2]); acc1[2] = wmma16(a1, b2, acc1[2]);
    acc0[3] = wmma16(a0, b3, acc0[3]); acc1[3] = wmma16(a1, b3, acc1[3]);
    a0 = a0n; a1 = a1n; b0 = b0n; b1 = b1n; b2 = b2n; b3 = b3n;
  }
  // last k-group
  acc0[0] = wmma16(a0, b0, acc0[0]); acc1[0] = wmma16(a1, b0, acc1[0]);
  acc0[1] = wmma16(a0, b1, acc0[1]); acc1[1] = wmma16(a1, b1, acc1[1]);
  acc0[2] = wmma16(a0, b2, acc0[2]); acc1[2] = wmma16(a1, b2, acc1[2]);
  acc0[3] = wmma16(a0, b3, acc0[3]); acc1[3] = wmma16(a1, b3, acc1[3]);

#pragma unroll
  for (int nb = 0; nb < 4; ++nb) {
    const int col = n0 + nb * 16 + ln;
    const float bv = bias ? bias[col] : 0.0f;
#pragma unroll
    for (int r = 0; r < 8; ++r) {
      const int row = m0 + r + (hi << 3);
      Out[(size_t)row * Nn + col] = acc0[nb][r] + bv;
      Out[(size_t)(row + 16) * Nn + col] = acc1[nb][r] + bv;
    }
  }
}

// ---------------- per-head norm + RoPE + layout shuffle ----------------
__global__ void qkv_post_kernel(const float* __restrict__ qkv,   // (B*N)x3072
                                const float* __restrict__ pos,   // B*N*2
                                const float* __restrict__ qk_scale,
                                _Float16* __restrict__ q16,      // (B*H*N)x64
                                _Float16* __restrict__ k16,      // (B*H*KPADT)x64
                                _Float16* __restrict__ vT16) {   // (B*H*64)xKPADT
  const int lane = threadIdx.x & 31;
  const int wave = threadIdx.x >> 5;
  const int unit = blockIdx.x * 8 + wave;     // over B*N*H
  const int h = unit & (HEADS - 1);
  const int bn = unit >> 4;
  const int b = bn >> 10, n = bn & (NTOK - 1);
  const float* row = qkv + (size_t)bn * (3 * DIMC);

  float q1 = row[h * 64 + lane],        q2 = row[h * 64 + lane + 32];
  float k1 = row[1024 + h * 64 + lane], k2 = row[1024 + h * 64 + lane + 32];
  float v1 = row[2048 + h * 64 + lane], v2 = row[2048 + h * 64 + lane + 32];

  float sq = q1 * q1 + q2 * q2, sk = k1 * k1 + k2 * k2;
#pragma unroll
  for (int m = 16; m >= 1; m >>= 1) {
    sq += __shfl_xor(sq, m, 32); sk += __shfl_xor(sk, m, 32);
  }
  const float s = __expf(0.5f * fminf(qk_scale[h], MAX_LOG_SCALE) - QLOG_HD);
  const float rq = s * rsqrtf(sq * (1.0f / 64.0f) + EPSV);
  const float rk = s * rsqrtf(sk * (1.0f / 64.0f) + EPSV);
  q1 *= rq; q2 *= rq; k1 *= rk; k2 *= rk;

  // RoPE: pair (j, j+32), angle = (j<16 ? px : py) * freq[h][j%16]
  const float px = pos[(size_t)bn * 2], py = pos[(size_t)bn * 2 + 1];
  const float fr = __expf(LOG_PI + (float)((lane & 15) * 16 + h) * FREQ_STEP);
  const float ang = (lane < 16 ? px : py) * fr;
  float sn, cs; __sincosf(ang, &sn, &cs);
  float nq1 = q1 * cs - q2 * sn, nq2 = q2 * cs + q1 * sn;
  float nk1 = k1 * cs - k2 * sn, nk2 = k2 * cs + k1 * sn;
  nq1 *= 0.125f; nq2 *= 0.125f;               // fold 1/sqrt(HD) into q

  const size_t bh = (size_t)b * HEADS + h;
  _Float16* qrow = q16 + (bh * NTOK + n) * 64;
  qrow[lane] = (_Float16)nq1; qrow[lane + 32] = (_Float16)nq2;
  _Float16* krow = k16 + (bh * KPADT + n) * 64;
  krow[lane] = (_Float16)nk1; krow[lane + 32] = (_Float16)nk2;
  _Float16* vcol = vT16 + bh * 64 * KPADT + n;
  vcol[(size_t)lane * KPADT] = (_Float16)v1;
  vcol[(size_t)(lane + 32) * KPADT] = (_Float16)v2;
}

__global__ void cond_post_kernel(const float* __restrict__ kcvc,  // (B*LPAD)x2048
                                 _Float16* __restrict__ k16,
                                 _Float16* __restrict__ vT16) {
  const int lane = threadIdx.x & 31;
  const int wave = threadIdx.x >> 5;
  const int unit = blockIdx.x * 8 + wave;     // over B*LPAD*H
  const int h = unit & (HEADS - 1);
  const int bl = unit >> 4;
  const int b = bl / LPAD, lc = bl % LPAD;
  const float* row = kcvc + (size_t)bl * (2 * DIMC);
  const float k1 = row[h * 64 + lane],        k2 = row[h * 64 + lane + 32];
  const float v1 = row[1024 + h * 64 + lane], v2 = row[1024 + h * 64 + lane + 32];
  const size_t bh = (size_t)b * HEADS + h;
  const int key = NTOK + lc;
  _Float16* krow = k16 + (bh * KPADT + key) * 64;
  krow[lane] = (_Float16)k1; krow[lane + 32] = (_Float16)k2;
  _Float16* vcol = vT16 + bh * 64 * KPADT + key;
  vcol[(size_t)lane * KPADT] = (_Float16)v1;
  vcol[(size_t)(lane + 32) * KPADT] = (_Float16)v2;
}

// ---------------- flash attention: 1 wave = 16 queries, KV chunks of 32 ------
__global__ void __launch_bounds__(128)
attn_kernel(const _Float16* __restrict__ q16, const _Float16* __restrict__ k16,
            const _Float16* __restrict__ vT16, const float* __restrict__ kvbias,
            _Float16* __restrict__ out16) {
  __shared__ _Float16 lds_p[4][16 * 32];      // per-wave 16x32 P tile
  const int lane = threadIdx.x & 31;
  const int wave = threadIdx.x >> 5;
  const int ln = lane & 15, hi = lane >> 4;
  const int bh = blockIdx.x >> 4;             // b*H + h
  const int qt = ((blockIdx.x & 15) << 2) + wave;   // 0..63
  const int q0 = qt << 4;

  const _Float16* qrow = q16 + ((size_t)bh * NTOK + q0 + ln) * 64;
  const v16h aq0 = load_a_frag(qrow, hi);        // d 0..31
  const v16h aq1 = load_a_frag(qrow + 32, hi);   // d 32..63
  const float* biasrow = kvbias + (size_t)(bh >> 4) * KPADT;
  const _Float16* kbh = k16 + (size_t)bh * KPADT * 64;
  const _Float16* vbh = vT16 + (size_t)bh * 64 * KPADT;

  v8f o0 = {}, o1 = {}, o2 = {}, o3 = {};
  float mrow[8], srow[8];
#pragma unroll
  for (int r = 0; r < 8; ++r) { mrow[r] = -1e30f; srow[r] = 0.f; }

  for (int kb = 0; kb < KPADT; kb += 32) {
    // issue ALL chunk loads up front: K frags then V frags; V loads complete
    // while the softmax VALU block runs.
    const _Float16* kr0 = kbh + (size_t)(kb + ln) * 64 + (hi << 4);
    const _Float16* kr1 = kbh + (size_t)(kb + 16 + ln) * 64 + (hi << 4);
    const v16h bk00 = *(const v16h*)kr0;
    const v16h bk01 = *(const v16h*)(kr0 + 32);
    const v16h bk10 = *(const v16h*)kr1;
    const v16h bk11 = *(const v16h*)(kr1 + 32);
    const _Float16* vp = vbh + (size_t)ln * KPADT + kb + (hi << 4);
    const v16h vb0 = *(const v16h*)(vp);
    const v16h vb1 = *(const v16h*)(vp + (size_t)16 * KPADT);
    const v16h vb2 = *(const v16h*)(vp + (size_t)32 * KPADT);
    const v16h vb3 = *(const v16h*)(vp + (size_t)48 * KPADT);
    __builtin_prefetch(kr0 + 32 * 64, 0, 1);   // next K chunk
    __builtin_prefetch(vp + 32, 0, 1);         // next V chunk

    v8f s0 = {}, s1 = {};
    s0 = wmma16(aq0, bk00, s0);
    s0 = wmma16(aq1, bk01, s0);
    s1 = wmma16(aq0, bk10, s1);
    s1 = wmma16(aq1, bk11, s1);
    const float bi0 = biasrow[kb + ln], bi1 = biasrow[kb + 16 + ln];

    // online softmax per row (rows live across 16-lane halves)
#pragma unroll
    for (int r = 0; r < 8; ++r) {
      float x0 = s0[r] + bi0, x1 = s1[r] + bi1;
      float mx = fmaxf(x0, x1);
#pragma unroll
      for (int msk = 8; msk >= 1; msk >>= 1) mx = fmaxf(mx, __shfl_xor(mx, msk, 32));
      const float mnew = fmaxf(mrow[r], mx);
      const float alpha = __expf(mrow[r] - mnew);
      const float p0 = __expf(x0 - mnew), p1 = __expf(x1 - mnew);
      float ps = p0 + p1;
#pragma unroll
      for (int msk = 8; msk >= 1; msk >>= 1) ps += __shfl_xor(ps, msk, 32);
      srow[r] = srow[r] * alpha + ps;
      mrow[r] = mnew;
      o0[r] *= alpha; o1[r] *= alpha; o2[r] *= alpha; o3[r] *= alpha;
      const int rowi = r + (hi << 3);
      lds_p[wave][rowi * 32 + ln] = (_Float16)p0;
      lds_p[wave][rowi * 32 + 16 + ln] = (_Float16)p1;
    }
    asm volatile("s_wait_dscnt 0" ::: "memory");   // DS RAW: P tile visible
    const v16h pa = load_a_frag(&lds_p[wave][ln * 32], hi);

    // O += P @ V  (V stored d-major -> contiguous B fragments, preloaded)
    o0 = wmma16(pa, vb0, o0);
    o1 = wmma16(pa, vb1, o1);
    o2 = wmma16(pa, vb2, o2);
    o3 = wmma16(pa, vb3, o3);
  }

  const int b = bh >> 4, h = bh & (HEADS - 1);
#pragma unroll
  for (int r = 0; r < 8; ++r) {
    const float inv = 1.0f / srow[r];
    const int row = q0 + r + (hi << 3);
    _Float16* orow = out16 + ((size_t)(b * NTOK) + row) * DIMC + h * 64 + ln;
    orow[0]  = (_Float16)(o0[r] * inv);
    orow[16] = (_Float16)(o1[r] * inv);
    orow[32] = (_Float16)(o2[r] * inv);
    orow[48] = (_Float16)(o3[r] * inv);
  }
}

// ---------------- host orchestration ----------------
extern "C" void kernel_launch(void* const* d_in, const int* in_sizes, int n_in,
                              void* d_out, int out_size, void* d_ws, size_t ws_size,
                              hipStream_t stream) {
  (void)in_sizes; (void)n_in; (void)out_size; (void)ws_size;
  const float* x        = (const float*)d_in[0];
  const float* cond     = (const float*)d_in[1];
  const unsigned char* cmask = (const unsigned char*)d_in[2];
  const float* pos      = (const float*)d_in[3];
  const float* nscale   = (const float*)d_in[4];
  const float* w_qkv    = (const float*)d_in[5];
  const float* qk_scale = (const float*)d_in[6];
  const float* w_kv     = (const float*)d_in[7];
  const float* w_dense  = (const float*)d_in[8];
  const float* b_dense  = (const float*)d_in[9];
  float* out = (float*)d_out;

  char* ws = (char*)d_ws;
  size_t off = 0;
  auto carve = [&](size_t bytes) -> char* {
    char* p = ws + off;
    off += (bytes + 255) & ~(size_t)255;
    return p;
  };
  _Float16* w_qkv16   = (_Float16*)carve((size_t)3 * DIMC * DIMC * 2);
  _Float16* w_kv16    = (_Float16*)carve((size_t)2 * DIMC * DIMC * 2);
  _Float16* w_dense16 = (_Float16*)carve((size_t)DIMC * DIMC * 2);
  _Float16* xn16      = (_Float16*)carve((size_t)BATCH * NTOK * DIMC * 2);
  _Float16* cond16    = (_Float16*)carve((size_t)BATCH * LPAD * DIMC * 2);
  float*    qkv32     = (float*)carve((size_t)BATCH * NTOK * 3 * DIMC * 4);
  float*    kcvc32    = (float*)carve((size_t)BATCH * LPAD * 2 * DIMC * 4);
  _Float16* q16       = (_Float16*)carve((size_t)BATCH * HEADS * NTOK * HD * 2);
  _Float16* k16       = (_Float16*)carve((size_t)BATCH * HEADS * KPADT * HD * 2);
  _Float16* vT16      = (_Float16*)carve((size_t)BATCH * HEADS * HD * KPADT * 2);
  float*    kvbias    = (float*)carve((size_t)BATCH * KPADT * 4);
  _Float16* attn16    = (_Float16*)carve((size_t)BATCH * NTOK * DIMC * 2);

  // 1) weight / input conversion
  long nwq = 3L * DIMC * DIMC;
  cvt_f32_f16_kernel<<<(int)((nwq + 255) / 256), 256, 0, stream>>>(w_qkv, w_qkv16, nwq);
  long nwk = 2L * DIMC * DIMC;
  cvt_f32_f16_kernel<<<(int)((nwk + 255) / 256), 256, 0, stream>>>(w_kv, w_kv16, nwk);
  long nwd = 1L * DIMC * DIMC;
  cvt_f32_f16_kernel<<<(int)((nwd + 255) / 256), 256, 0, stream>>>(w_dense, w_dense16, nwd);
  long nc = (long)BATCH * LPAD * DIMC;
  build_cond16_kernel<<<(int)((nc + 255) / 256), 256, 0, stream>>>(cond, cond16);
  rmsnorm_x_kernel<<<BATCH * NTOK, 256, 0, stream>>>(x, nscale, xn16);
  fill_bias_kernel<<<(BATCH * KPADT + 255) / 256, 256, 0, stream>>>(cmask, kvbias);

  // 2) projections (32x64 wave tiles)
  gemm_wmma_kernel<<<(BATCH * NTOK / 32) * (3 * DIMC / 64), 32, 0, stream>>>(
      xn16, w_qkv16, qkv32, nullptr, BATCH * NTOK, 3 * DIMC, DIMC);
  gemm_wmma_kernel<<<(BATCH * LPAD / 32) * (2 * DIMC / 64), 32, 0, stream>>>(
      cond16, w_kv16, kcvc32, nullptr, BATCH * LPAD, 2 * DIMC, DIMC);

  // 3) per-head norm + rope + layout
  qkv_post_kernel<<<BATCH * NTOK * HEADS / 8, 256, 0, stream>>>(
      qkv32, pos, qk_scale, q16, k16, vT16);
  cond_post_kernel<<<BATCH * LPAD * HEADS / 8, 256, 0, stream>>>(kcvc32, k16, vT16);

  // 4) flash attention
  attn_kernel<<<BATCH * HEADS * (NTOK / 64), 128, 0, stream>>>(
      q16, k16, vT16, kvbias, attn16);

  // 5) output projection + bias
  gemm_wmma_kernel<<<(BATCH * NTOK / 32) * (DIMC / 64), 32, 0, stream>>>(
      attn16, w_dense16, out, b_dense, BATCH * NTOK, DIMC, DIMC);
}
